// MoEClassifier_68667937129034
// MI455X (gfx1250) — compile-verified, matrix operand order checked by reading, not run on hardware
//
#include <hip/hip_runtime.h>
#include <hip/hip_bf16.h>
#include <math.h>

// ---------------------------------------------------------------------------
// MoE classifier forward for MI455X (gfx1250, wave32, WMMA).
//  * all big GEMMs on v_wmma_f32_16x16x32_bf16 (f32 accumulate)
//  * weights converted+transposed to bf16 once per launch (L2-resident)
//  * LDS tiles staged with GLOBAL_LOAD_ASYNC_TO_LDS_B128 (ASYNCcnt) when the
//    toolchain exposes it, double-buffered; sync-copy fallback otherwise
//  * patch-mean hoisted before the encoder's 2nd linear layer (linearity)
// ---------------------------------------------------------------------------

typedef __attribute__((ext_vector_type(16))) __bf16 v16bf;
typedef __attribute__((ext_vector_type(8)))  __bf16 v8bf;
typedef __attribute__((ext_vector_type(8)))  float  v8f;
typedef __attribute__((ext_vector_type(4)))  int    v4i;

#define BM 64
#define BN 128
#define KC 32
#define LDS_STRIDE 48   // 96B row stride: keeps 16B alignment, staggers banks

#if __has_builtin(__builtin_amdgcn_global_load_async_to_lds_b128)
#define HAVE_ASYNC_COPY 1
#else
#define HAVE_ASYNC_COPY 0
#endif
#if __has_builtin(__builtin_amdgcn_s_wait_asynccnt)
#define HAVE_ASYNC_WAIT 1
#else
#define HAVE_ASYNC_WAIT 0
#endif

// 16-byte global -> LDS copy (async when available, tracked by ASYNCcnt)
__device__ __forceinline__ void copy16(const __bf16* __restrict__ g, __bf16* l) {
#if HAVE_ASYNC_COPY
  __builtin_amdgcn_global_load_async_to_lds_b128((v4i*)g, (v4i*)l, 0, 0);
#else
  *(uint4*)l = *(const uint4*)g;
#endif
}

template<int N>
__device__ __forceinline__ void wait_async() {
#if HAVE_ASYNC_WAIT
  __builtin_amdgcn_s_wait_asynccnt(N);
#elif HAVE_ASYNC_COPY
  asm volatile("s_wait_asynccnt %0" :: "i"(N) : "memory");
#endif
}

__device__ __forceinline__ v16bf cat16(v8bf lo, v8bf hi) {
  return __builtin_shufflevector(lo, hi, 0,1,2,3,4,5,6,7,8,9,10,11,12,13,14,15);
}

// Wt[n][k] = (bf16) W[k][n]; rows n in [N, Npad) zero-filled.
// Grid: (Npad/32, K/32), block 256 (32x8), LDS-tiled for coalescing.
__global__ __launch_bounds__(256) void transpose_convert_kernel(
    const float* __restrict__ W, __bf16* __restrict__ Wt,
    int K, int N, int Npad)
{
  __shared__ float tile[32][33];
  const int tx = threadIdx.x & 31;
  const int ty = threadIdx.x >> 5;
  const int n0 = blockIdx.x * 32;
  const int k0 = blockIdx.y * 32;
  #pragma unroll
  for (int i = 0; i < 4; ++i) {
    const int n = n0 + tx;
    tile[ty + i * 8][tx] = (n < N) ? W[(size_t)(k0 + ty + i * 8) * N + n] : 0.f;
  }
  __syncthreads();
  #pragma unroll
  for (int i = 0; i < 4; ++i) {
    const int n = n0 + ty + i * 8;
    Wt[(size_t)n * K + (k0 + tx)] = (__bf16)tile[tx][ty + i * 8];
  }
}

// C[m0:m0+64, n0:n0+128] = act(A @ W + bias)
//   A : (M,K) row-major, f32 (A_F32, converted while staging) or bf16, lda=K
//   Wt: (Npad,K) row-major bf16 = transposed weight, stride K
//   C : bf16 or f32 (OUT_F32), leading dim ldc; stores guarded by col<N
// Block 256 = 8 waves; wave w owns col-tile w (16 cols) x 4 row-tiles of 16.
// Double-buffered LDS tiles, async-to-LDS staging pipelined one step ahead.
template<bool A_F32, bool RELU, bool OUT_F32>
__global__ __launch_bounds__(256) void gemm_bias_act_wmma(
    const void* __restrict__ Av, int lda,
    const __bf16* __restrict__ Wt, int N,
    const float* __restrict__ bias,
    void* __restrict__ Cv, int ldc, int K)
{
  __shared__ __bf16 As[2][BM][LDS_STRIDE];
  __shared__ __bf16 Bs[2][BN][LDS_STRIDE];

  const int tid  = threadIdx.x;
  const int w    = tid >> 5;
  const int lane = tid & 31;
  const int m0   = blockIdx.x * BM;
  const int n0   = blockIdx.y * BN;

  v8f zero = {0.f,0.f,0.f,0.f,0.f,0.f,0.f,0.f};
  v8f acc[4] = {zero, zero, zero, zero};

  // staging coords: A tile 64x32 bf16 (256 thr x 16B), B tile 128x32 (x2)
  const int arow = tid >> 2;
  const int acol = (tid & 3) * 8;

  auto stage = [&](int k0, int buf) {
    if (A_F32) {
      const float* A = (const float*)Av;
      const float* ap = A + (size_t)(m0 + arow) * lda + (k0 + acol);
      const float4* p = (const float4*)ap;
      float4 f0 = p[0], f1 = p[1];
      v8bf vb;
      vb[0]=(__bf16)f0.x; vb[1]=(__bf16)f0.y; vb[2]=(__bf16)f0.z; vb[3]=(__bf16)f0.w;
      vb[4]=(__bf16)f1.x; vb[5]=(__bf16)f1.y; vb[6]=(__bf16)f1.z; vb[7]=(__bf16)f1.w;
      *(v8bf*)&As[buf][arow][acol] = vb;
      if (k0 + KC < K) __builtin_prefetch(ap + KC, 0, 1);   // global_prefetch_b8
    } else {
      const __bf16* A = (const __bf16*)Av;
      copy16(A + (size_t)(m0 + arow) * lda + (k0 + acol), &As[buf][arow][acol]);
    }
    #pragma unroll
    for (int q = 0; q < 2; ++q) {
      const int id = tid + 256 * q;
      const int brow = id >> 2, bcol = (id & 3) * 8;
      copy16(Wt + (size_t)(n0 + brow) * K + (k0 + bcol), &Bs[buf][brow][bcol]);
    }
  };

  stage(0, 0);
  int buf = 0;
  for (int k0 = 0; k0 < K; k0 += KC) {
    const bool last = (k0 + KC >= K);
    if (!last) stage(k0 + KC, buf ^ 1);          // issue next tile ahead
    if (last) wait_async<0>();                    // drain
    else      wait_async<A_F32 ? 2 : 3>();        // current tile done, next in flight
    __syncthreads();

    // ISA 16-bit fragment layout: lanes 0-15 hold K 0-7 & 16-23,
    // lanes 16-31 hold K 8-15 & 24-31 -> two aligned ds_load_b128 each.
    const int half = lane >> 4;
    const int l16  = lane & 15;
    const int koff = half * 8;
    const int bc   = w * 16 + l16;
    v8bf blo = *(const v8bf*)&Bs[buf][bc][koff];
    v8bf bhi = *(const v8bf*)&Bs[buf][bc][koff + 16];
    v16bf bfrag = cat16(blo, bhi);
    #pragma unroll
    for (int mt = 0; mt < 4; ++mt) {
      const int ar = mt * 16 + l16;
      v8bf alo = *(const v8bf*)&As[buf][ar][koff];
      v8bf ahi = *(const v8bf*)&As[buf][ar][koff + 16];
      v16bf afrag = cat16(alo, ahi);
      acc[mt] = __builtin_amdgcn_wmma_f32_16x16x32_bf16(
          false, afrag, false, bfrag, (short)0, acc[mt], false, false);
    }
    __syncthreads();
    buf ^= 1;
  }

  // epilogue: C layout = VGPR r -> (M=r | M=8+r), N = lane&15
  const int half = lane >> 4;
  const int col  = n0 + w * 16 + (lane & 15);
  const float bv = (col < N) ? bias[col] : 0.f;
  #pragma unroll
  for (int mt = 0; mt < 4; ++mt) {
    #pragma unroll
    for (int r = 0; r < 8; ++r) {
      float v = acc[mt][r] + bv;
      if (RELU) v = fmaxf(v, 0.f);
      const int m = m0 + mt * 16 + half * 8 + r;
      if (col < N) {
        if (OUT_F32) ((float*)Cv)[(size_t)m * ldc + col] = v;
        else         ((__bf16*)Cv)[(size_t)m * ldc + col] = (__bf16)v;
      }
    }
  }
}

// hbar[b,:] = mean over 9 patches of h[b*9+p,:]   (bf16 in/out, f32 accum)
__global__ __launch_bounds__(256) void patch_mean_kernel(
    const __bf16* __restrict__ h, __bf16* __restrict__ hbar)
{
  const int id = blockIdx.x * 256 + threadIdx.x;   // 8192*512 threads
  const int b = id >> 9, j = id & 511;
  const __bf16* p = h + (size_t)b * 9 * 512 + j;
  float s = 0.f;
  #pragma unroll
  for (int pp = 0; pp < 9; ++pp) s += (float)p[pp * 512];
  hbar[id] = (__bf16)(s * (1.0f / 9.0f));
}

// One wave32 per row: neural logits (K=1024,E=8) via shuffle reduction,
// symbolic MLP (6->64->8) on lane 0, then softmax -> gate[b,8].
__global__ __launch_bounds__(256) void gating_softmax_kernel(
    const __bf16* __restrict__ g1, const float* __restrict__ gn_w2,
    const float* __restrict__ gn_b2, const float* __restrict__ sym,
    const float* __restrict__ gs_w1, const float* __restrict__ gs_b1,
    const float* __restrict__ gs_w2, const float* __restrict__ gs_b2,
    float* __restrict__ gate)
{
  const int lane = threadIdx.x & 31;
  const int b    = blockIdx.x * 8 + (threadIdx.x >> 5);
  float acc[8] = {0,0,0,0,0,0,0,0};
  const __bf16* row = g1 + (size_t)b * 1024;
  for (int k = lane; k < 1024; k += 32) {
    float x = (float)row[k];
    const float* wr = gn_w2 + (size_t)k * 8;
    #pragma unroll
    for (int e = 0; e < 8; ++e) acc[e] = fmaf(x, wr[e], acc[e]);
  }
  #pragma unroll
  for (int off = 16; off > 0; off >>= 1) {
    #pragma unroll
    for (int e = 0; e < 8; ++e) acc[e] += __shfl_down(acc[e], off, 32);
  }
  if (lane == 0) {
    float so[8];
    #pragma unroll
    for (int e = 0; e < 8; ++e) so[e] = gs_b2[e];
    float sf[6];
    #pragma unroll
    for (int s = 0; s < 6; ++s) sf[s] = sym[b * 6 + s];
    for (int hI = 0; hI < 64; ++hI) {
      float hv = gs_b1[hI];
      #pragma unroll
      for (int s = 0; s < 6; ++s) hv = fmaf(sf[s], gs_w1[s * 64 + hI], hv);
      hv = fmaxf(hv, 0.f);
      #pragma unroll
      for (int e = 0; e < 8; ++e) so[e] = fmaf(hv, gs_w2[hI * 8 + e], so[e]);
    }
    float lg[8], mx = -1e30f;
    #pragma unroll
    for (int e = 0; e < 8; ++e) {
      lg[e] = 0.5f * (acc[e] + gn_b2[e]) + 0.5f * so[e];
      mx = fmaxf(mx, lg[e]);
    }
    float sum = 0.f;
    #pragma unroll
    for (int e = 0; e < 8; ++e) { lg[e] = __expf(lg[e] - mx); sum += lg[e]; }
    const float inv = 1.f / sum;
    #pragma unroll
    for (int e = 0; e < 8; ++e) gate[(size_t)b * 8 + e] = lg[e] * inv;
  }
}

// logits[b,c] = sum_e gate[b,e] * expert_logits[b,e,c]
__global__ __launch_bounds__(256) void gate_reduce_kernel(
    const float* __restrict__ gate, const float* __restrict__ elog,
    float* __restrict__ out)
{
  const int id = blockIdx.x * 256 + threadIdx.x;   // 8192*1000 threads
  const int b = id / 1000, c = id - b * 1000;
  const float* g = gate + (size_t)b * 8;
  const float* e = elog + (size_t)b * 8000 + c;
  float s = 0.f;
  #pragma unroll
  for (int k = 0; k < 8; ++k) s = fmaf(g[k], e[(size_t)k * 1000], s);
  out[id] = s;
}

extern "C" void kernel_launch(void* const* d_in, const int* in_sizes, int n_in,
                              void* d_out, int out_size, void* d_ws, size_t ws_size,
                              hipStream_t stream) {
  const float* patches = (const float*)d_in[0];   // (73728,768)
  const float* sym     = (const float*)d_in[1];   // (8192,6)
  const float* enc_w1  = (const float*)d_in[2];   // (768,512)
  const float* enc_b1  = (const float*)d_in[3];
  const float* enc_w2  = (const float*)d_in[4];   // (512,1024)
  const float* enc_b2  = (const float*)d_in[5];
  const float* gn_w1   = (const float*)d_in[6];   // (1024,1024)
  const float* gn_b1   = (const float*)d_in[7];
  const float* gn_w2   = (const float*)d_in[8];   // (1024,8)
  const float* gn_b2   = (const float*)d_in[9];
  const float* gs_w1   = (const float*)d_in[10];  // (6,64)
  const float* gs_b1   = (const float*)d_in[11];
  const float* gs_w2   = (const float*)d_in[12];  // (64,8)
  const float* gs_b2   = (const float*)d_in[13];
  const float* ew1     = (const float*)d_in[14];  // (8,1024,2048)
  const float* eb1     = (const float*)d_in[15];  // (8,2048)
  const float* ew2     = (const float*)d_in[16];  // (8,2048,1000)
  const float* eb2     = (const float*)d_in[17];  // (8,1000)

  // outputs: logits (8192,1000) | gate (8192,8) | expert_logits (8192,8,1000)
  float* logits = (float*)d_out;
  float* gate   = logits + (size_t)8192 * 1000;
  float* elog   = gate   + (size_t)8192 * 8;

  // workspace (~222 MB)
  char* ws = (char*)d_ws;
  __bf16* h_bf    = (__bf16*)ws; ws += (size_t)73728 * 512  * 2;  // enc hidden
  __bf16* hbar_bf = (__bf16*)ws; ws += (size_t)8192  * 512  * 2;  // patch-mean hidden
  __bf16* agg_bf  = (__bf16*)ws; ws += (size_t)8192  * 1024 * 2;  // agg embedding
  __bf16* g1_bf   = (__bf16*)ws; ws += (size_t)8192  * 1024 * 2;  // gating hidden
  __bf16* eh_bf   = (__bf16*)ws; ws += (size_t)8192  * 2048 * 2;  // expert hidden
  __bf16* enc_w1t = (__bf16*)ws; ws += (size_t)512  * 768  * 2;   // bf16 W^T
  __bf16* enc_w2t = (__bf16*)ws; ws += (size_t)1024 * 512  * 2;
  __bf16* gn_w1t  = (__bf16*)ws; ws += (size_t)1024 * 1024 * 2;
  __bf16* ew1t    = (__bf16*)ws; ws += (size_t)8 * 2048 * 1024 * 2;
  __bf16* ew2t    = (__bf16*)ws; ws += (size_t)8 * 1024 * 2048 * 2; // Npad=1024

  const dim3 blk(256);

  // 0) one-time weight convert+transpose to bf16 (weights then L2-resident)
  transpose_convert_kernel<<<dim3(512 / 32, 768 / 32), blk, 0, stream>>>(
      enc_w1, enc_w1t, 768, 512, 512);
  transpose_convert_kernel<<<dim3(1024 / 32, 512 / 32), blk, 0, stream>>>(
      enc_w2, enc_w2t, 512, 1024, 1024);
  transpose_convert_kernel<<<dim3(1024 / 32, 1024 / 32), blk, 0, stream>>>(
      gn_w1, gn_w1t, 1024, 1024, 1024);
  for (int e = 0; e < 8; ++e) {
    transpose_convert_kernel<<<dim3(2048 / 32, 1024 / 32), blk, 0, stream>>>(
        ew1 + (size_t)e * 1024 * 2048, ew1t + (size_t)e * 2048 * 1024,
        1024, 2048, 2048);
    transpose_convert_kernel<<<dim3(1024 / 32, 2048 / 32), blk, 0, stream>>>(
        ew2 + (size_t)e * 2048 * 1000, ew2t + (size_t)e * 1024 * 2048,
        2048, 1000, 1024);
  }

  // 1) encoder L1: h = relu(X @ enc_w1 + b1)
  gemm_bias_act_wmma<true, true, false>
      <<<dim3(73728 / BM, 512 / BN), blk, 0, stream>>>(
          patches, 768, enc_w1t, 512, enc_b1, h_bf, 512, 768);

  // 2) mean over patches (hoisted before linear layer)
  patch_mean_kernel<<<dim3((8192 * 512) / 256), blk, 0, stream>>>(h_bf, hbar_bf);

  // 3) agg = hbar @ enc_w2 + b2
  gemm_bias_act_wmma<false, false, false>
      <<<dim3(8192 / BM, 1024 / BN), blk, 0, stream>>>(
          hbar_bf, 512, enc_w2t, 1024, enc_b2, agg_bf, 1024, 512);

  // 4) gating hidden: g1 = relu(agg @ gn_w1 + gn_b1)
  gemm_bias_act_wmma<false, true, false>
      <<<dim3(8192 / BM, 1024 / BN), blk, 0, stream>>>(
          agg_bf, 1024, gn_w1t, 1024, gn_b1, g1_bf, 1024, 1024);

  // 5) gate = softmax(0.5*neural + 0.5*symbolic)
  gating_softmax_kernel<<<dim3(8192 / 8), blk, 0, stream>>>(
      g1_bf, gn_w2, gn_b2, sym, gs_w1, gs_b1, gs_w2, gs_b2, gate);

  // 6) experts (sequential over e; hidden buffer reused)
  for (int e = 0; e < 8; ++e) {
    gemm_bias_act_wmma<false, true, false>
        <<<dim3(8192 / BM, 2048 / BN), blk, 0, stream>>>(
            agg_bf, 1024, ew1t + (size_t)e * 2048 * 1024, 2048,
            eb1 + (size_t)e * 2048, eh_bf, 2048, 1024);
    gemm_bias_act_wmma<false, false, true>
        <<<dim3(8192 / BM, (1000 + BN - 1) / BN), blk, 0, stream>>>(
            eh_bf, 2048, ew2t + (size_t)e * 1024 * 2048, 1000,
            eb2 + (size_t)e * 1000, elog + (size_t)e * 1000, 8000, 2048);
  }

  // 7) logits = gate-weighted sum of expert logits
  gate_reduce_kernel<<<dim3((8192 * 1000) / 256), blk, 0, stream>>>(
      gate, elog, logits);
}
